// BiDirectionalCrossAttention_90288802496621
// MI455X (gfx1250) — matrix operand order repaired and checked by reading, not compiled, not run on hardware
//
#include <hip/hip_runtime.h>
#include <math.h>

typedef __bf16 bf16_t;
typedef __attribute__((ext_vector_type(16))) __bf16 v16bf;
typedef __attribute__((ext_vector_type(8)))  __bf16 v8bf;
typedef __attribute__((ext_vector_type(8)))  float   v8f;
typedef __attribute__((ext_vector_type(4)))  int     v4i;

#define TM 128
#define TN 128
#define TKK 32
#define LDS_PAD 8
#define LDK (TKK + LDS_PAD)

#if defined(__gfx1250__) && \
    __has_builtin(__builtin_amdgcn_global_load_async_to_lds_b128) && \
    __has_builtin(__builtin_amdgcn_s_wait_asynccnt)
#define USE_ASYNC 1
typedef v4i __attribute__((address_space(1)))* gptr_v4i;
typedef v4i __attribute__((address_space(3)))* lptr_v4i;
#else
#define USE_ASYNC 0
#endif

// ---------------------------------------------------------------- helpers ---

__global__ __launch_bounds__(256)
void cvt_f32_bf16(const float* __restrict__ in, bf16_t* __restrict__ out, long n) {
  long i = (long)blockIdx.x * 256 + threadIdx.x;
  if (i < n) out[i] = (bf16_t)in[i];
}

// conv_w (512,256,5) -> Wc (512,1280) with Wc[e][k*256+c] = conv_w[e][c][k]
__global__ __launch_bounds__(256)
void repack_conv_w(const float* __restrict__ w, bf16_t* __restrict__ out) {
  int i = blockIdx.x * 256 + threadIdx.x;     // 512*1280 exact
  int e = i / 1280, r = i - e * 1280;
  int k = r >> 8, c = r & 255;
  out[i] = (bf16_t)w[(e * 256 + c) * 5 + k];
}

// issue one 128x32 bf16 tile (rows x K) into an LDS tile, 16B chunks.
// src points at tile origin, ldk = source row stride (elements).
__device__ __forceinline__
void issue_tile(const bf16_t* __restrict__ src, int ldk, bf16_t* dst, int tid) {
#pragma unroll
  for (int i = 0; i < 2; ++i) {
    int c   = tid + i * 256;   // 512 chunks of 8 bf16
    int row = c >> 2;
    int kc  = (c & 3) << 3;
#if USE_ASYNC
    __builtin_amdgcn_global_load_async_to_lds_b128(
        (gptr_v4i)(src + (size_t)row * ldk + kc),
        (lptr_v4i)(dst + row * LDK + kc),
        0, 0);
#else
    *(uint4*)(dst + row * LDK + kc) = *(const uint4*)(src + (size_t)row * ldk + kc);
#endif
  }
}

template <int N>
__device__ __forceinline__ void async_wait() {
#if USE_ASYNC
  __builtin_amdgcn_s_wait_asynccnt(N);
#endif
}

// ------------------------------------------------------------ WMMA GEMM ----
// C[b] = alpha * A[b](MxK) @ B[b]^T  (+bias[N]) (+base[b])
// A: row-major bf16 (M,K).  B: row-major bf16 (N,K)  [always "weights" layout]
// Cf: optional f32 out (M,N).  Cb: optional bf16 out; if cb_trans, stored (N,M).
__global__ __launch_bounds__(256)
void gemm_bf16_wmma(const bf16_t* __restrict__ A, long long sAb,
                    const bf16_t* __restrict__ Bm, long long sBb,
                    float* __restrict__ Cf, bf16_t* __restrict__ Cb, long long sCb,
                    const float* __restrict__ bias,
                    const float* __restrict__ base, long long sBaseb,
                    int M, int N, int K, float alpha, int cb_trans) {
  __shared__ bf16_t As[2][TM * LDK];
  __shared__ bf16_t Bs[2][TN * LDK];

  const int tid = threadIdx.x;
  const int bz  = blockIdx.z;
  A  += (size_t)bz * sAb;
  Bm += (size_t)bz * sBb;
  const size_t cOff = (size_t)bz * sCb;

  const int n0 = blockIdx.x * TN;
  const int m0 = blockIdx.y * TM;

  const int lane = tid & 31;
  const int hl   = lane & 15;
  const int hi   = lane >> 4;
  const int wave = tid >> 5;   // 0..7
  const int wm   = wave & 3;   // row group: rows wm*32
  const int wn   = wave >> 2;  // col group: cols wn*64

  const bf16_t* Abase = A  + (size_t)m0 * K;
  const bf16_t* Bbase = Bm + (size_t)n0 * K;

  v8f acc[2][4] = {};

  const int nt = K / TKK;
  issue_tile(Abase, K, As[0], tid);
  issue_tile(Bbase, K, Bs[0], tid);

  for (int kt = 0; kt < nt; ++kt) {
    const int cur = kt & 1;
    if (kt + 1 < nt) {
      // prefetch next K-tile into the other buffer (4 async ops / wave)
      issue_tile(Abase + (size_t)(kt + 1) * TKK, K, As[cur ^ 1], tid);
      issue_tile(Bbase + (size_t)(kt + 1) * TKK, K, Bs[cur ^ 1], tid);
      async_wait<4>();   // current tile's 4 ops complete; next 4 in flight
    } else {
      async_wait<0>();
    }
    __syncthreads();     // all waves' tile data visible

    union F16x16 { v16bf v; v8bf h[2]; };
    F16x16 af[2], bfr[4];
    const bf16_t* Asb = As[cur];
    const bf16_t* Bsb = Bs[cur];
#pragma unroll
    for (int mi = 0; mi < 2; ++mi) {
      const int ar = wm * 32 + mi * 16 + hl;
      // A 16x32 frag: lane-half hi holds K in [8*hi,8*hi+8) and [16+8*hi, +8)
      af[mi].h[0] = *(const v8bf*)(Asb + ar * LDK + hi * 8);
      af[mi].h[1] = *(const v8bf*)(Asb + ar * LDK + 16 + hi * 8);
    }
#pragma unroll
    for (int ni = 0; ni < 4; ++ni) {
      const int br = wn * 64 + ni * 16 + hl;
      // B 32x16 frag: lane-half hi holds K in [8*hi,8*hi+8) and [16+8*hi, +8)
      bfr[ni].h[0] = *(const v8bf*)(Bsb + br * LDK + hi * 8);
      bfr[ni].h[1] = *(const v8bf*)(Bsb + br * LDK + 16 + hi * 8);
    }
#pragma unroll
    for (int mi = 0; mi < 2; ++mi)
#pragma unroll
      for (int ni = 0; ni < 4; ++ni)
        acc[mi][ni] = __builtin_amdgcn_wmma_f32_16x16x32_bf16(
            false, af[mi].v, false, bfr[ni].v, (short)0, acc[mi][ni], false, false);
    __syncthreads();     // protect buffer reuse before next prefetch
  }

  // epilogue: C 16x16 layout -> element r is row (hi*8 + r), col hl
#pragma unroll
  for (int mi = 0; mi < 2; ++mi) {
#pragma unroll
    for (int ni = 0; ni < 4; ++ni) {
      const int row0 = m0 + wm * 32 + mi * 16 + hi * 8;
      const int col  = n0 + wn * 64 + ni * 16 + hl;
      if (cb_trans) {
        // bf16 output stored transposed (N,M): 8 consecutive rows -> 16B store
        __align__(16) bf16_t t[8];
#pragma unroll
        for (int r = 0; r < 8; ++r) t[r] = (bf16_t)(acc[mi][ni][r] * alpha);
        *(uint4*)&Cb[cOff + (size_t)col * M + row0] = *(const uint4*)t;
      } else {
        const float bcol = bias ? bias[col] : 0.0f;
#pragma unroll
        for (int r = 0; r < 8; ++r) {
          const size_t idx = (size_t)(row0 + r) * N + col;
          float v = acc[mi][ni][r] * alpha + bcol;
          if (base) v += base[(size_t)bz * sBaseb + idx];
          if (Cf) Cf[cOff + idx] = v;
          if (Cb) Cb[cOff + idx] = (bf16_t)v;
        }
      }
    }
  }
}

// ----------------------------------------------------------- softmaxes -----

// one 256-thread block per (b,l) row of 2048; P = softmax over s (bf16)
__global__ __launch_bounds__(256)
void row_softmax_bf16(const float* __restrict__ Ar, bf16_t* __restrict__ P) {
  const int tid = threadIdx.x;
  const size_t rbase = ((size_t)blockIdx.y * gridDim.x + blockIdx.x) * 2048;
  float v[8];
  float m = -3.4e38f;
#pragma unroll
  for (int i = 0; i < 8; ++i) { v[i] = Ar[rbase + tid + i * 256]; m = fmaxf(m, v[i]); }
  __shared__ float red[256];
  red[tid] = m; __syncthreads();
  for (int o = 128; o > 0; o >>= 1) { if (tid < o) red[tid] = fmaxf(red[tid], red[tid + o]); __syncthreads(); }
  m = red[0]; __syncthreads();
  float d = 0.0f;
#pragma unroll
  for (int i = 0; i < 8; ++i) { v[i] = __expf(v[i] - m); d += v[i]; }
  red[tid] = d; __syncthreads();
  for (int o = 128; o > 0; o >>= 1) { if (tid < o) red[tid] += red[tid + o]; __syncthreads(); }
  const float rd = 1.0f / red[0];
#pragma unroll
  for (int i = 0; i < 8; ++i) P[rbase + tid + i * 256] = (bf16_t)(v[i] * rd);
}

// column softmax over l with online max/sum; writes PT[b][s][l] (transposed)
__global__ __launch_bounds__(256)
void col_softmax_T_bf16(const float* __restrict__ Ar, bf16_t* __restrict__ PT,
                        int Lr, int Sc) {
  const int tid = threadIdx.x;
  const int s = blockIdx.x * 256 + tid;
  const size_t aB = (size_t)blockIdx.y * Lr * Sc;
  const size_t pB = (size_t)blockIdx.y * Sc * Lr;
  float m = -3.4e38f, d = 0.0f;
  for (int l = 0; l < Lr; ++l) {
    const float a = Ar[aB + (size_t)l * Sc + s];
    const float mn = fmaxf(m, a);
    d = d * __expf(m - mn) + __expf(a - mn);
    m = mn;
  }
  const float rd = 1.0f / d;
  const size_t prow = pB + (size_t)s * Lr;
  for (int l0 = 0; l0 < Lr; l0 += 16) {
    __align__(16) bf16_t tmp[16];
#pragma unroll
    for (int ll = 0; ll < 16; ++ll)
      tmp[ll] = (bf16_t)(__expf(Ar[aB + (size_t)(l0 + ll) * Sc + s] - m) * rd);
    *(uint4*)&PT[prow + l0]     = *(const uint4*)&tmp[0];
    *(uint4*)&PT[prow + l0 + 8] = *(const uint4*)&tmp[8];
  }
}

// out[b] = concat(upd_lat[b] (1024,512), upd_tok[b] (2048,512))
__global__ __launch_bounds__(256)
void concat_out(const float* __restrict__ ul, const float* __restrict__ ut,
                float* __restrict__ out) {
  const size_t i = (size_t)blockIdx.x * 256 + threadIdx.x;
  const size_t perB = (size_t)3072 * 512;
  const size_t b = i / perB, r = i - b * perB;
  const size_t row = r >> 9, col = r & 511;
  out[i] = (row < 1024) ? ul[(b * 1024 + row) * 512 + col]
                        : ut[(b * 2048 + (row - 1024)) * 512 + col];
}

// ---------------------------------------------------------------- launch ---

extern "C" void kernel_launch(void* const* d_in, const int* in_sizes, int n_in,
                              void* d_out, int out_size, void* d_ws, size_t ws_size,
                              hipStream_t stream) {
  (void)in_sizes; (void)n_in; (void)out_size; (void)ws_size;
  const int B = 8, L = 1024, S10 = 10240, S = 2048, D = 512, TD = 256, Kc = 1280;

  const float* latents = (const float*)d_in[0];
  const float* tokens  = (const float*)d_in[1];
  const float* W_lat   = (const float*)d_in[2];
  const float* W_tok   = (const float*)d_in[3];
  const float* W_vlat  = (const float*)d_in[4];
  const float* W_vtok  = (const float*)d_in[5];
  const float* conv_w  = (const float*)d_in[6];
  const float* conv_b  = (const float*)d_in[7];

  float* out     = (float*)d_out;
  float* upd_lat = out;
  float* upd_tok = out + (size_t)B * L * D;
  float* out_cat = upd_tok + (size_t)B * S * D;

  char* ws = (char*)d_ws;
  size_t off = 0;
  auto alloc = [&](size_t bytes) -> void* {
    void* p = ws + off;
    off += (bytes + 255) & ~(size_t)255;
    return p;
  };
  bf16_t* tokens_h  = (bf16_t*)alloc((size_t)B * S10 * TD * 2);
  bf16_t* latents_h = (bf16_t*)alloc((size_t)B * L * D * 2);
  bf16_t* Wc_h      = (bf16_t*)alloc((size_t)D * Kc * 2);
  bf16_t* Wlat_h    = (bf16_t*)alloc((size_t)D * D * 2);
  bf16_t* Wtok_h    = (bf16_t*)alloc((size_t)D * D * 2);
  bf16_t* Wvlat_h   = (bf16_t*)alloc((size_t)D * D * 2);
  bf16_t* Wvtok_h   = (bf16_t*)alloc((size_t)D * D * 2);
  float*  tok_f     = (float*)alloc((size_t)B * S * D * 4);
  bf16_t* tok_h     = (bf16_t*)alloc((size_t)B * S * D * 2);
  bf16_t* Rlat_h    = (bf16_t*)alloc((size_t)B * L * D * 2);
  bf16_t* Rtok_h    = (bf16_t*)alloc((size_t)B * S * D * 2);
  bf16_t* VlatT_h   = (bf16_t*)alloc((size_t)B * L * D * 2);  // (D, L) per batch
  bf16_t* VtokT_h   = (bf16_t*)alloc((size_t)B * S * D * 2);  // (D, S) per batch
  float*  Asc       = (float*)alloc((size_t)B * L * S * 4);
  bf16_t* Prow      = (bf16_t*)alloc((size_t)B * L * S * 2);
  bf16_t* PcolT     = (bf16_t*)alloc((size_t)B * S * L * 2);

  const float inv_sqrt_d = 1.0f / sqrtf((float)D);

  // 1) bf16 casts + weight repacks
  {
    long n = (long)B * S10 * TD;
    cvt_f32_bf16<<<(n + 255) / 256, 256, 0, stream>>>(tokens, tokens_h, n);
  }
  {
    long n = (long)B * L * D;
    cvt_f32_bf16<<<(n + 255) / 256, 256, 0, stream>>>(latents, latents_h, n);
  }
  {
    long n = (long)D * D;
    cvt_f32_bf16<<<(n + 255) / 256, 256, 0, stream>>>(W_lat,  Wlat_h,  n);
    cvt_f32_bf16<<<(n + 255) / 256, 256, 0, stream>>>(W_tok,  Wtok_h,  n);
    cvt_f32_bf16<<<(n + 255) / 256, 256, 0, stream>>>(W_vlat, Wvlat_h, n);
    cvt_f32_bf16<<<(n + 255) / 256, 256, 0, stream>>>(W_vtok, Wvtok_h, n);
  }
  repack_conv_w<<<(D * Kc) / 256, 256, 0, stream>>>(conv_w, Wc_h);

  // 2) conv-as-GEMM: (B*2048, 1280) @ Wc^T + bias -> tok (f32 + bf16)
  gemm_bf16_wmma<<<dim3(D / TN, (B * S) / TM, 1), 256, 0, stream>>>(
      tokens_h, 0, Wc_h, 0, tok_f, tok_h, 0, conv_b, nullptr, 0,
      B * S, D, Kc, 1.0f, 0);

  // 3) projections: R (row-major bf16), V (transposed bf16 -> (D, seq))
  gemm_bf16_wmma<<<dim3(D / TN, L / TM, B), 256, 0, stream>>>(
      latents_h, (long long)L * D, Wlat_h, 0, nullptr, Rlat_h, (long long)L * D,
      nullptr, nullptr, 0, L, D, D, 1.0f, 0);
  gemm_bf16_wmma<<<dim3(D / TN, L / TM, B), 256, 0, stream>>>(
      latents_h, (long long)L * D, Wvlat_h, 0, nullptr, VlatT_h, (long long)L * D,
      nullptr, nullptr, 0, L, D, D, 1.0f, 1);
  gemm_bf16_wmma<<<dim3(D / TN, S / TM, B), 256, 0, stream>>>(
      tok_h, (long long)S * D, Wtok_h, 0, nullptr, Rtok_h, (long long)S * D,
      nullptr, nullptr, 0, S, D, D, 1.0f, 0);
  gemm_bf16_wmma<<<dim3(D / TN, S / TM, B), 256, 0, stream>>>(
      tok_h, (long long)S * D, Wvtok_h, 0, nullptr, VtokT_h, (long long)S * D,
      nullptr, nullptr, 0, S, D, D, 1.0f, 1);

  // 4) scores A = R_lat @ R_tok^T / sqrt(D)  (f32)
  gemm_bf16_wmma<<<dim3(S / TN, L / TM, B), 256, 0, stream>>>(
      Rlat_h, (long long)L * D, Rtok_h, (long long)S * D, Asc, nullptr,
      (long long)L * S, nullptr, nullptr, 0, L, S, D, inv_sqrt_d, 0);

  // 5) softmaxes
  row_softmax_bf16<<<dim3(L, B), 256, 0, stream>>>(Asc, Prow);
  col_softmax_T_bf16<<<dim3(S / 256, B), 256, 0, stream>>>(Asc, PcolT, L, S);

  // 6) delta_lat: P_row (L,S) @ VtokT(512,S)^T + latents -> upd_lat
  gemm_bf16_wmma<<<dim3(D / TN, L / TM, B), 256, 0, stream>>>(
      Prow, (long long)L * S, VtokT_h, (long long)S * D, upd_lat, nullptr,
      (long long)L * D, nullptr, latents, (long long)L * D, L, D, S, 1.0f, 0);

  // 7) delta_tok: P_colT (S,L) @ VlatT(512,L)^T + tok -> upd_tok
  gemm_bf16_wmma<<<dim3(D / TN, S / TM, B), 256, 0, stream>>>(
      PcolT, (long long)S * L, VlatT_h, (long long)L * D, upd_tok, nullptr,
      (long long)S * D, nullptr, tok_f, (long long)S * D, S, D, L, 1.0f, 0);

  // 8) concat
  concat_out<<<((size_t)B * 3072 * 512) / 256, 256, 0, stream>>>(
      upd_lat, upd_tok, out_cat);
}
